// MultiScaleAttention_79242146611849
// MI455X (gfx1250) — matrix operand order repaired and checked by reading, not compiled, not run on hardware
//
#include <hip/hip_runtime.h>
#include <hip/hip_bf16.h>
#include <stdint.h>

// ---------------------------------------------------------------------------
// MultiScaleAttention (MViT pooled attention) for MI455X / gfx1250.
// GEMMs on v_wmma_f32_16x16x32_bf16 with LDS-staged, double-buffered tiles
// fed by global_load_async_to_lds_b128 (ASYNCcnt path). wave32 layouts per
// CDNA5 ISA 7.12.2.
// ---------------------------------------------------------------------------

typedef __attribute__((ext_vector_type(16))) __bf16 v16bf;
typedef __attribute__((ext_vector_type(8)))  float  v8f;

#define WMMA_BF16(A, Bv, C) \
  __builtin_amdgcn_wmma_f32_16x16x32_bf16(false, (A), false, (Bv), (short)0, (C), false, false)

#define BATCH     8
#define NHEADS    12
#define HEADDIM   64
#define DMODEL    768
#define NQ        1568      // 8*14*14
#define NKEYS     392       // 8*7*7
#define SCALE_F   0.125f    // 64^-0.5

// ---------------------------------------------------------------------------
// CDNA5 async copy helpers (ASYNCcnt-tracked memory->LDS DMA)
// ---------------------------------------------------------------------------
__device__ __forceinline__ void async_ld_b128(const void* lds_ptr, const void* gptr) {
  uint32_t l = (uint32_t)(uintptr_t)lds_ptr;          // flat LDS addr: low 32 bits = LDS offset
  unsigned long long g = (unsigned long long)(uintptr_t)gptr;
  asm volatile("global_load_async_to_lds_b128 %0, %1, off"
               :: "v"(l), "v"(g) : "memory");
}
__device__ __forceinline__ void wait_async0() {
  asm volatile("s_wait_asynccnt 0x0" ::: "memory");
}

// ---------------------------------------------------------------------------
// fp32 -> bf16 elementwise convert
// ---------------------------------------------------------------------------
__global__ void cvt_f32_to_bf16(const float* __restrict__ in,
                                __bf16* __restrict__ out, int n) {
  int i = blockIdx.x * blockDim.x + threadIdx.x;
  if (i < n) out[i] = (__bf16)in[i];
}

// ---------------------------------------------------------------------------
// LDS fragment loads (contiguous -> ds_load_b128)
// A frag (16x32 MxK): lane m = lane&15, half=lane>>4; element e -> k=e+16*half.
//                     Abuf row-major [row][32].
// B frag (32x16 KxN): lane n = lane&15; element e -> k=e+16*half. Bt is [n][32].
// ---------------------------------------------------------------------------
__device__ __forceinline__ v16bf frag_from_A(const __bf16* __restrict__ Abuf,
                                             int row, int half) {
  const __bf16* r = Abuf + row * 32;
  v16bf a;
#pragma unroll
  for (int j = 0; j < 8; ++j) {
    a[j]     = r[16 * half + j];
    a[8 + j] = r[16 * half + 8 + j];
  }
  return a;
}
__device__ __forceinline__ v16bf frag_from_Bt(const __bf16* __restrict__ Bt,
                                              int n, int half) {
  const __bf16* r = Bt + n * 32 + 16 * half;
  v16bf b;
#pragma unroll
  for (int j = 0; j < 16; ++j) b[j] = r[j];
  return b;
}

// ---------------------------------------------------------------------------
// Tile staging. Block = 256 threads. Block tile M=128, N=64, Kchunk=32.
// A tile: 128x32 bf16 = 8KB, async DMA (512 b128 transfers, 2 per thread).
// B tile: staged transposed [n][k] 64x32 via coalesced b128 read + b16 scatter.
// ---------------------------------------------------------------------------
__device__ __forceinline__ void stage_A_async(const __bf16* __restrict__ A, int lda,
                                              int mblock, int kbase,
                                              __bf16* __restrict__ Abuf) {
  int t = threadIdx.x;
#pragma unroll
  for (int u = 0; u < 2; ++u) {
    int idx = t * 2 + u;
    int row = idx >> 2, seg = idx & 3;
    async_ld_b128(Abuf + row * 32 + seg * 8,
                  A + (size_t)(mblock + row) * lda + kbase + seg * 8);
  }
}
__device__ __forceinline__ void stage_B_tr(const __bf16* __restrict__ W, int ldb,
                                           int nblock, int kbase,
                                           __bf16* __restrict__ Bt) {
  int t = threadIdx.x;
  int k = t >> 3, nseg = t & 7;
  const __bf16* g = W + (size_t)(kbase + k) * ldb + nblock + nseg * 8;
  __bf16 tmp[8];
#pragma unroll
  for (int j = 0; j < 8; ++j) tmp[j] = g[j];
#pragma unroll
  for (int j = 0; j < 8; ++j) Bt[(nseg * 8 + j) * 32 + k] = tmp[j];
}

// Double-buffered K pipeline producing 2x2 fragment accumulators per wave.
__device__ __forceinline__ void gemm_core(const __bf16* __restrict__ A, int lda,
                                          const __bf16* __restrict__ W, int ldb,
                                          int mblock, int nblock, int nK,
                                          __bf16* __restrict__ Abuf,   // [2][128*32]
                                          __bf16* __restrict__ Btb,    // [2][64*32]
                                          v8f acc[2][2]) {
  int wave = threadIdx.x >> 5, lane = threadIdx.x & 31;
  int mlo = lane & 15, half = lane >> 4;
  int wm = (wave >> 1) * 32, wn = (wave & 1) * 32;

  stage_A_async(A, lda, mblock, 0, Abuf);
  stage_B_tr(W, ldb, nblock, 0, Btb);
  wait_async0();
  __syncthreads();

  for (int kb = 0; kb < nK; ++kb) {
    int cur = kb & 1;
    __bf16* Ac = Abuf + cur * (128 * 32);
    __bf16* Bc = Btb + cur * (64 * 32);

    if (kb + 1 < nK) {   // prefetch next chunk into the other buffer
      stage_A_async(A, lda, mblock, (kb + 1) * 32, Abuf + (cur ^ 1) * (128 * 32));
      stage_B_tr(W, ldb, nblock, (kb + 1) * 32, Btb + (cur ^ 1) * (64 * 32));
    }

    v16bf a0 = frag_from_A(Ac, wm + mlo, half);
    v16bf a1 = frag_from_A(Ac, wm + 16 + mlo, half);
    v16bf b0 = frag_from_Bt(Bc, wn + mlo, half);
    v16bf b1 = frag_from_Bt(Bc, wn + 16 + mlo, half);
    acc[0][0] = WMMA_BF16(a0, b0, acc[0][0]);
    acc[0][1] = WMMA_BF16(a0, b1, acc[0][1]);
    acc[1][0] = WMMA_BF16(a1, b0, acc[1][0]);
    acc[1][1] = WMMA_BF16(a1, b1, acc[1][1]);

    if (kb + 1 < nK) wait_async0();
    __syncthreads();
  }
}

// ---------------------------------------------------------------------------
// QKV GEMM: [12544 x 768] @ [768 x 2304] + bias -> scatter q/k/v [B,nh,N,64].
// Grid: (12544/128) * (2304/64) = 98 * 36 = 3528 blocks x 256.
// ---------------------------------------------------------------------------
__global__ __launch_bounds__(256) void qkv_gemm_kernel(
    const __bf16* __restrict__ A, const __bf16* __restrict__ W,
    const float* __restrict__ bias,
    float* __restrict__ qraw, float* __restrict__ kraw, float* __restrict__ vraw) {
  __shared__ __bf16 Abuf[2][128 * 32];
  __shared__ __bf16 Btb[2][64 * 32];

  int nb = blockIdx.x % 36, mb = blockIdx.x / 36;
  int mblock = mb * 128, nblock = nb * 64;

  v8f acc[2][2] = {};
  gemm_core(A, DMODEL, W, 3 * DMODEL, mblock, nblock, 24, &Abuf[0][0], &Btb[0][0], acc);

  int wave = threadIdx.x >> 5, lane = threadIdx.x & 31;
  int mlo = lane & 15, half = lane >> 4;
  int wm = (wave >> 1) * 32, wn = (wave & 1) * 32;
#pragma unroll
  for (int i = 0; i < 2; ++i) {
#pragma unroll
    for (int j = 0; j < 2; ++j) {
      int ncol  = nblock + wn + j * 16 + mlo;
      int which = ncol / DMODEL;
      int rest  = ncol % DMODEL;
      int head  = rest >> 6;
      int ch    = rest & 63;
      float* dst = (which == 0) ? qraw : ((which == 1) ? kraw : vraw);
      float bv = bias[ncol];
#pragma unroll
      for (int r = 0; r < 8; ++r) {
        int mrow = mblock + wm + i * 16 + r + 8 * half;
        int bidx = mrow / NQ, pos = mrow % NQ;
        dst[(((size_t)bidx * NHEADS + head) * NQ + pos) * HEADDIM + ch] = acc[i][j][r] + bv;
      }
    }
  }
}

// ---------------------------------------------------------------------------
// Proj GEMM: [12544 x 768] @ [768 x 768] + bias -> fp32 d_out.
// Grid: 98 * 12 = 1176 blocks x 256.
// ---------------------------------------------------------------------------
__global__ __launch_bounds__(256) void proj_gemm_kernel(
    const __bf16* __restrict__ A, const __bf16* __restrict__ W,
    const float* __restrict__ bias, float* __restrict__ out) {
  __shared__ __bf16 Abuf[2][128 * 32];
  __shared__ __bf16 Btb[2][64 * 32];

  int nb = blockIdx.x % 12, mb = blockIdx.x / 12;
  int mblock = mb * 128, nblock = nb * 64;

  v8f acc[2][2] = {};
  gemm_core(A, DMODEL, W, DMODEL, mblock, nblock, 24, &Abuf[0][0], &Btb[0][0], acc);

  int wave = threadIdx.x >> 5, lane = threadIdx.x & 31;
  int mlo = lane & 15, half = lane >> 4;
  int wm = (wave >> 1) * 32, wn = (wave & 1) * 32;
#pragma unroll
  for (int i = 0; i < 2; ++i) {
#pragma unroll
    for (int j = 0; j < 2; ++j) {
      int ncol = nblock + wn + j * 16 + mlo;
      float bv = bias[ncol];
#pragma unroll
      for (int r = 0; r < 8; ++r) {
        int mrow = mblock + wm + i * 16 + r + 8 * half;
        out[(size_t)mrow * DMODEL + ncol] = acc[i][j][r] + bv;
      }
    }
  }
}

// ---------------------------------------------------------------------------
// Fused depthwise conv3d (3x3x3, pad 1) + LayerNorm over 64 channels.
// ---------------------------------------------------------------------------
__global__ __launch_bounds__(64) void pool_ln_kernel(
    const float* __restrict__ in,   // [B*nh][1568][64]
    const float* __restrict__ pw,   // [64][27]
    const float* __restrict__ gam, const float* __restrict__ bet,
    float* __restrict__ outf,       // may be null
    __bf16* __restrict__ outb,      // may be null
    int sh, int OH, int OW) {
  int Lout = 8 * OH * OW;
  int bh = blockIdx.x / Lout;
  int l  = blockIdx.x % Lout;
  int ot = l / (OH * OW);
  int rem = l % (OH * OW);
  int oy = rem / OW, ox = rem % OW;
  int c = threadIdx.x;

  const float* base = in + (size_t)bh * NQ * HEADDIM;
  float s = 0.f;
#pragma unroll
  for (int dt = 0; dt < 3; ++dt) {
    int it = ot + dt - 1;
    if (it < 0 || it >= 8) continue;
#pragma unroll
    for (int dy = 0; dy < 3; ++dy) {
      int iy = oy * sh + dy - 1;
      if (iy < 0 || iy >= 14) continue;
#pragma unroll
      for (int dx = 0; dx < 3; ++dx) {
        int ix = ox * sh + dx - 1;
        if (ix < 0 || ix >= 14) continue;
        s += base[((size_t)it * 196 + iy * 14 + ix) * HEADDIM + c] *
             pw[c * 27 + dt * 9 + dy * 3 + dx];
      }
    }
  }

  __shared__ float red[64];
  red[c] = s;
  __syncthreads();
  for (int off = 32; off > 0; off >>= 1) {
    if (c < off) red[c] += red[c + off];
    __syncthreads();
  }
  float mean = red[0] * (1.0f / 64.0f);
  __syncthreads();
  float d = s - mean;
  red[c] = d * d;
  __syncthreads();
  for (int off = 32; off > 0; off >>= 1) {
    if (c < off) red[c] += red[c + off];
    __syncthreads();
  }
  float var = red[0] * (1.0f / 64.0f);
  float y = d * rsqrtf(var + 1e-5f) * gam[c] + bet[c];

  size_t oidx = ((size_t)bh * Lout + l) * HEADDIM + c;
  if (outf) outf[oidx] = y;
  if (outb) outb[oidx] = (__bf16)y;
}

// ---------------------------------------------------------------------------
// Fused attention: one block per (b, head, 16-query tile). 128 threads / 4 waves.
// ---------------------------------------------------------------------------
__global__ __launch_bounds__(128) void attn_kernel(
    const float* __restrict__ qpool,  // [B*nh][1568][64] fp32 (post-LN, unscaled)
    const __bf16* __restrict__ kbf,   // [B*nh][392][64]
    const __bf16* __restrict__ vbf,   // [B*nh][392][64]
    const float* __restrict__ relh, const float* __restrict__ relw,
    const float* __restrict__ relt,
    __bf16* __restrict__ attnout)     // [B][1568][768]
{
  __shared__ float  S[16][400];
  __shared__ __bf16 P[16][416];
  __shared__ float  rps[3][16][8];   // [0]=h(kh), [1]=w(kw), [2]=t(kt)
  __shared__ float  red[16][8];
  __shared__ float  rowstat[16];

  int tid = threadIdx.x;
  int qt  = blockIdx.x % 98;
  int bh  = blockIdx.x / 98;
  int hh  = bh % NHEADS;
  int bb  = bh / NHEADS;
  int qt0 = qt * 16;

  // relative-position bias dot products (22 entries per query row)
  for (int tsk = tid; tsk < 16 * 22; tsk += 128) {
    int row = tsk / 22, e = tsk % 22;
    int qpos = qt0 + row;
    int tq = qpos / 196, qr = qpos % 196;
    int yq = qr / 14, xq = qr % 14;
    const float* qrow = qpool + ((size_t)bh * NQ + qpos) * HEADDIM;
    const float* rrow;
    int which, idx2;
    if (e < 7)       { which = 0; idx2 = e;      rrow = relh + (size_t)(yq - 2 * idx2 + 12) * HEADDIM; }
    else if (e < 14) { which = 1; idx2 = e - 7;  rrow = relw + (size_t)(xq - 2 * idx2 + 12) * HEADDIM; }
    else             { which = 2; idx2 = e - 14; rrow = relt + (size_t)(tq - idx2 + 7) * HEADDIM; }
    float s = 0.f;
#pragma unroll 8
    for (int c = 0; c < HEADDIM; ++c) s += qrow[c] * rrow[c];
    rps[which][row][idx2] = s;
  }
  __syncthreads();

  int wave = tid >> 5, lane = tid & 31;
  int mlo = lane & 15, half = lane >> 4;

  // q fragments (two 16x32 chunks over the 64 channels), scaled
  const float* qrow = qpool + ((size_t)bh * NQ + qt0 + mlo) * HEADDIM;
  v16bf aq0, aq1;
#pragma unroll
  for (int p = 0; p < 8; ++p) {
    int k0 = 2 * p + 16 * half;
    aq0[2 * p]     = (__bf16)(qrow[k0] * SCALE_F);
    aq0[2 * p + 1] = (__bf16)(qrow[k0 + 1] * SCALE_F);
    aq1[2 * p]     = (__bf16)(qrow[k0 + 32] * SCALE_F);
    aq1[2 * p + 1] = (__bf16)(qrow[k0 + 33] * SCALE_F);
  }

  // scores: 25 key tiles of 16 (tile 24 partially masked)
  const __bf16* kb = kbf + (size_t)bh * NKEYS * HEADDIM;
  for (int ktile = wave; ktile < 25; ktile += 4) {
    int key = ktile * 16 + mlo;
    bool valid = key < NKEYS;
    v16bf b0 = {}, b1 = {};
    if (valid) {
      const __bf16* krow = kb + (size_t)key * HEADDIM + 16 * half;
#pragma unroll
      for (int p = 0; p < 8; ++p) {
        b0[2 * p]     = krow[2 * p];
        b0[2 * p + 1] = krow[2 * p + 1];
        b1[2 * p]     = krow[2 * p + 32];
        b1[2 * p + 1] = krow[2 * p + 33];
      }
    }
    v8f c = {};
    c = WMMA_BF16(aq0, b0, c);
    c = WMMA_BF16(aq1, b1, c);

    int kt = key / 49, krem = key % 49;
    int kh = krem / 7, kw = krem % 7;
#pragma unroll
    for (int r = 0; r < 8; ++r) {
      int mm = r + 8 * half;
      float v = valid ? (c[r] + rps[0][mm][kh] + rps[1][mm][kw] + rps[2][mm][kt])
                      : -1e30f;
      S[mm][ktile * 16 + mlo] = v;
    }
  }
  __syncthreads();

  // softmax over 392 keys: 8 threads per row, 49 cols each
  {
    int row = tid >> 3, seg = tid & 7;
    int c0 = seg * 49;
    float mx = -1e30f;
    for (int i = 0; i < 49; ++i) mx = fmaxf(mx, S[row][c0 + i]);
    red[row][seg] = mx;
    __syncthreads();
    if (seg == 0) {
      float m2 = red[row][0];
#pragma unroll
      for (int j = 1; j < 8; ++j) m2 = fmaxf(m2, red[row][j]);
      rowstat[row] = m2;
    }
    __syncthreads();
    float rm = rowstat[row];
    float s = 0.f;
    for (int i = 0; i < 49; ++i) {
      float e = __expf(S[row][c0 + i] - rm);
      P[row][c0 + i] = (__bf16)e;
      s += e;
    }
    red[row][seg] = s;
    __syncthreads();
    if (seg == 0) {
      float t = 0.f;
#pragma unroll
      for (int j = 0; j < 8; ++j) t += red[row][j];
      rowstat[row] = 1.0f / t;
    }
    __syncthreads();
    float inv = rowstat[row];
    for (int i = 0; i < 49; ++i)
      P[row][c0 + i] = (__bf16)((float)P[row][c0 + i] * inv);
    if (tid < 16) {
      for (int c = NKEYS; c < 416; ++c) P[tid][c] = (__bf16)0.f;
    }
  }
  __syncthreads();

  // out = P[16 x 416] @ V[416 x 64]; wave owns 16 output channels
  const __bf16* vb = vbf + (size_t)bh * NKEYS * HEADDIM;
  int nc = wave * 16 + mlo;
  v8f o = {};
  for (int kc = 0; kc < 13; ++kc) {
    v16bf a, bv;
#pragma unroll
    for (int p = 0; p < 8; ++p) {
      int k0 = 2 * p + 16 * half + 32 * kc;
      a[2 * p]     = P[mlo][k0];
      a[2 * p + 1] = P[mlo][k0 + 1];
      int key0 = 2 * p + 16 * half + 32 * kc;
      bv[2 * p]     = (key0     < NKEYS) ? vb[(size_t)key0 * HEADDIM + nc]       : (__bf16)0.f;
      bv[2 * p + 1] = (key0 + 1 < NKEYS) ? vb[(size_t)(key0 + 1) * HEADDIM + nc] : (__bf16)0.f;
    }
    o = WMMA_BF16(a, bv, o);
  }
#pragma unroll
  for (int r = 0; r < 8; ++r) {
    int mm = r + 8 * half;
    int qpos = qt0 + mm;
    attnout[((size_t)bb * NQ + qpos) * DMODEL + hh * HEADDIM + nc] = (__bf16)o[r];
  }
}

// ---------------------------------------------------------------------------
// Host launcher
// ---------------------------------------------------------------------------
extern "C" void kernel_launch(void* const* d_in, const int* in_sizes, int n_in,
                              void* d_out, int out_size, void* d_ws, size_t ws_size,
                              hipStream_t stream) {
  (void)in_sizes; (void)n_in; (void)out_size; (void)ws_size;

  const float* x      = (const float*)d_in[0];
  const float* w_qkv  = (const float*)d_in[1];
  const float* b_qkv  = (const float*)d_in[2];
  const float* pqw    = (const float*)d_in[3];
  const float* pkw    = (const float*)d_in[4];
  const float* pvw    = (const float*)d_in[5];
  const float* nqg    = (const float*)d_in[6];
  const float* nqb    = (const float*)d_in[7];
  const float* nkg    = (const float*)d_in[8];
  const float* nkb    = (const float*)d_in[9];
  const float* nvg    = (const float*)d_in[10];
  const float* nvb    = (const float*)d_in[11];
  const float* relh   = (const float*)d_in[12];
  const float* relw   = (const float*)d_in[13];
  const float* relt   = (const float*)d_in[14];
  const float* w_proj = (const float*)d_in[15];
  const float* b_proj = (const float*)d_in[16];
  float* out = (float*)d_out;

  char* ws = (char*)d_ws;
  size_t off = 0;
  auto alloc = [&](size_t bytes) -> char* {
    char* p = ws + off;
    off = (off + bytes + 255) & ~(size_t)255;
    return p;
  };

  const int NX    = BATCH * NQ * DMODEL;          // 9,633,792
  const int NWQKV = DMODEL * 3 * DMODEL;          // 1,769,472
  const int NWPRJ = DMODEL * DMODEL;              //   589,824
  const size_t RAW = (size_t)BATCH * NHEADS * NQ * HEADDIM;     // 9,633,792
  const size_t KVN = (size_t)BATCH * NHEADS * NKEYS * HEADDIM;  // 2,408,448

  __bf16* xbf     = (__bf16*)alloc((size_t)NX * 2);
  __bf16* wqkvbf  = (__bf16*)alloc((size_t)NWQKV * 2);
  __bf16* wprojbf = (__bf16*)alloc((size_t)NWPRJ * 2);
  float*  qraw    = (float*)alloc(RAW * 4);
  float*  kraw    = (float*)alloc(RAW * 4);
  float*  vraw    = (float*)alloc(RAW * 4);
  float*  qpool   = (float*)alloc(RAW * 4);
  __bf16* kbfp    = (__bf16*)alloc(KVN * 2);
  __bf16* vbfp    = (__bf16*)alloc(KVN * 2);
  __bf16* attnout = (__bf16*)alloc((size_t)NX * 2);

  // 1) bf16 conversions
  cvt_f32_to_bf16<<<(NX + 255) / 256, 256, 0, stream>>>(x, xbf, NX);
  cvt_f32_to_bf16<<<(NWQKV + 255) / 256, 256, 0, stream>>>(w_qkv, wqkvbf, NWQKV);
  cvt_f32_to_bf16<<<(NWPRJ + 255) / 256, 256, 0, stream>>>(w_proj, wprojbf, NWPRJ);

  // 2) QKV GEMM (98*36 block tiles)
  qkv_gemm_kernel<<<3528, 256, 0, stream>>>(xbf, wqkvbf, b_qkv, qraw, kraw, vraw);

  // 3) pooling + LayerNorm
  pool_ln_kernel<<<BATCH * NHEADS * NQ, 64, 0, stream>>>(
      qraw, pqw, nqg, nqb, qpool, (__bf16*)nullptr, 1, 14, 14);
  pool_ln_kernel<<<BATCH * NHEADS * NKEYS, 64, 0, stream>>>(
      kraw, pkw, nkg, nkb, (float*)nullptr, kbfp, 2, 7, 7);
  pool_ln_kernel<<<BATCH * NHEADS * NKEYS, 64, 0, stream>>>(
      vraw, pvw, nvg, nvb, (float*)nullptr, vbfp, 2, 7, 7);

  // 4) fused attention
  attn_kernel<<<BATCH * NHEADS * 98, 128, 0, stream>>>(
      qpool, kbfp, vbfp, relh, relw, relt, attnout);

  // 5) output projection (98*12 block tiles)
  proj_gemm_kernel<<<1176, 256, 0, stream>>>(attnout, wprojbf, b_proj, out);
}
